// MegaExpertRouter_46007689675028
// MI455X (gfx1250) — matrix operand order repaired and checked by reading, not compiled, not run on hardware
//
#include <hip/hip_runtime.h>
#include <math.h>

// ---------------------------------------------------------------------------
// MoE forward for MI455X (gfx1250): sparse top-8 routing + bf16 WMMA FFN.
// Weight-bandwidth bound: 536 MB of f32 expert weights streamed ~2x,
// converted to bf16 in registers and fed to v_wmma_f32_16x16x32_bf16.
// x is converted to bf16 once, then staged into LDS with
// GLOBAL_LOAD_ASYNC_TO_LDS_B128 (ASYNCcnt) in the GEMM kernels.
// ---------------------------------------------------------------------------

typedef __attribute__((ext_vector_type(16))) __bf16 v16bf;
typedef __attribute__((ext_vector_type(8)))  float  v8f;

#define T_TOK   2048
#define DMODEL  512
#define HGATE   1024
#define NEXP    64
#define TOPK    8
#define DFFN    2048
#define CAP     1024      // max tokens captured per expert (mean 256, sd ~16)
#define MTILE   128       // token tile (M)
#define SLICE   128       // FFN-column slice processed per iteration

union Frag16 { v16bf v; unsigned short u[16]; uint4 q[2]; };

__device__ __forceinline__ unsigned short f2bf(float f) {
  unsigned int u = __builtin_bit_cast(unsigned int, f);
  u += 0x7fffu + ((u >> 16) & 1u);          // round-to-nearest-even
  return (unsigned short)(u >> 16);
}
__device__ __forceinline__ float bf2f(unsigned short s) {
  unsigned int u = ((unsigned int)s) << 16;
  return __builtin_bit_cast(float, u);
}
__device__ __forceinline__ float gelu_exact(float x) {
  return 0.5f * x * (1.0f + erff(x * 0.70710678118654752f));
}

// ---- async global->LDS copy of one 16-byte chunk (per lane) ---------------
// VFLAT async form (ISA 15.18.3): vdst = LDS byte address, vaddr = 64-bit
// global address, "off" = no SADDR. Tracked by ASYNCcnt.
__device__ __forceinline__ void async_copy_b128(const unsigned short* gsrc,
                                                unsigned short* lds_dst) {
  unsigned int       loff  = (unsigned int)(unsigned long long)lds_dst; // LDS byte offset
  unsigned long long gaddr = (unsigned long long)gsrc;
  asm volatile("global_load_async_to_lds_b128 %0, %1, off"
               :: "v"(loff), "v"(gaddr) : "memory");
}
__device__ __forceinline__ void wait_async0() {
#if __has_builtin(__builtin_amdgcn_s_wait_asynccnt)
  __builtin_amdgcn_s_wait_asynccnt(0);
#else
  asm volatile("s_wait_asynccnt 0x0" ::: "memory");
#endif
}

// A fragment (16x32 bf16): lane m = lane&15; K base = ks + 8*(lane>=16);
// two contiguous 16-byte chunks at +0 and +16 halfwords.
__device__ __forceinline__ Frag16 load_a_lds(const unsigned short* base,
                                             int row, int ldk, int ks,
                                             int half) {
  Frag16 a;
  const unsigned short* p = base + (size_t)row * ldk + ks + (half ? 8 : 0);
  a.q[0] = *reinterpret_cast<const uint4*>(p);
  a.q[1] = *reinterpret_cast<const uint4*>(p + 16);
  return a;
}

// B fragment (32x16 bf16) from row-major f32 weights W[k][n] (ld = ldn):
// lane n = lane&15; K base = ks + 8*(lane>=16); u[j] = W[kb+j+(j>=8?8:0)][n].
__device__ __forceinline__ Frag16 load_b_f32(const float* __restrict__ w,
                                             int ks, int ldn, int n, int half) {
  Frag16 b;
  const float* p = w + (size_t)(ks + (half ? 8 : 0)) * ldn + n;
#pragma unroll
  for (int j = 0; j < 16; ++j) b.u[j] = f2bf(p[(size_t)(j + (j >= 8 ? 8 : 0)) * ldn]);
  return b;
}

// ---------------------------------------------------------------------------
__global__ void moe_zero_f32(float* __restrict__ p, int n) {
  int i = blockIdx.x * blockDim.x + threadIdx.x;
  if (i < n) p[i] = 0.0f;
}

// Convert x to bf16 once (read 4 MB, write 2 MB).
__global__ void moe_x_to_bf16(const float* __restrict__ x,
                              unsigned short* __restrict__ xbf, int n) {
  int i = blockIdx.x * blockDim.x + threadIdx.x;
  if (i < n) xbf[i] = f2bf(x[i]);
}

// ---------------------------------------------------------------------------
// Gate GEMM1: h = relu(x @ gw1 + gb1), bf16 output. grid=(16,8), block=256.
__global__ void moe_gate1_wmma(const unsigned short* __restrict__ xbf,
                               const float* __restrict__ gw1,
                               const float* __restrict__ gb1,
                               unsigned short* __restrict__ hout) {
  extern __shared__ char smem[];
  unsigned short* xs = (unsigned short*)smem;          // [128][512] bf16
  const int t0   = blockIdx.x * MTILE;
  const int wave = threadIdx.x >> 5;
  const int lane = threadIdx.x & 31;
  const int half = lane >> 4;
  const int lcol = lane & 15;

  // Async-stage the (contiguous) 128x512 bf16 tile: 8192 16-byte chunks.
  for (int c = threadIdx.x; c < MTILE * DMODEL / 8; c += blockDim.x)
    async_copy_b128(xbf + (size_t)t0 * DMODEL + c * 8, xs + c * 8);
  wait_async0();
  __syncthreads();

  const int ncol = blockIdx.y * SLICE + wave * 16 + lcol;
  v8f acc[8];
#pragma unroll
  for (int m = 0; m < 8; ++m) acc[m] = (v8f){0, 0, 0, 0, 0, 0, 0, 0};

  for (int ks = 0; ks < DMODEL; ks += 32) {
    Frag16 b = load_b_f32(gw1, ks, HGATE, ncol, half);
#pragma unroll
    for (int m = 0; m < 8; ++m) {
      Frag16 a = load_a_lds(xs, m * 16 + lcol, DMODEL, ks, half);
      acc[m] = __builtin_amdgcn_wmma_f32_16x16x32_bf16(
          false, a.v, false, b.v, (short)0, acc[m], false, false);
    }
  }

  const float bias = gb1[ncol];
#pragma unroll
  for (int m = 0; m < 8; ++m) {
#pragma unroll
    for (int r = 0; r < 8; ++r) {
      int row = m * 16 + (half ? r + 8 : r);
      float v = acc[m][r] + bias;
      v = v > 0.0f ? v : 0.0f;
      hout[(size_t)(t0 + row) * HGATE + ncol] = f2bf(v);
    }
  }
}

// ---------------------------------------------------------------------------
// Gate GEMM2 + softmax + top-8 + renormalize + usage accumulation.
// grid = T_TOK blocks of 64 threads (thread = expert).
__global__ void moe_gate2(const unsigned short* __restrict__ h,
                          const float* __restrict__ gw2,
                          const float* __restrict__ gb2,
                          int* __restrict__ topi, float* __restrict__ topv,
                          float* __restrict__ usage) {
  const int t = blockIdx.x;
  const int e = threadIdx.x;
  const unsigned short* hr = h + (size_t)t * HGATE;

  float acc = gb2[e];
  for (int k = 0; k < HGATE; ++k)
    acc += bf2f(hr[k]) * gw2[(size_t)k * NEXP + e];

  __shared__ float sc[NEXP];
  __shared__ float p[NEXP];
  sc[e] = acc;
  __syncthreads();

  if (e == 0) {
    float mx = sc[0];
    for (int i = 1; i < NEXP; ++i) mx = fmaxf(mx, sc[i]);
    float sum = 0.0f;
    for (int i = 0; i < NEXP; ++i) { float ex = __expf(sc[i] - mx); p[i] = ex; sum += ex; }
    float inv = 1.0f / sum;
    for (int i = 0; i < NEXP; ++i) p[i] *= inv;
    // top-8, first-index wins on ties (matches lax.top_k)
    bool used[NEXP];
    for (int i = 0; i < NEXP; ++i) used[i] = false;
    int   ti[TOPK]; float tv[TOPK]; float tsum = 0.0f;
    for (int k = 0; k < TOPK; ++k) {
      int bi = 0; float bv = -1.0f;
      for (int i = 0; i < NEXP; ++i)
        if (!used[i] && p[i] > bv) { bv = p[i]; bi = i; }
      used[bi] = true; ti[k] = bi; tv[k] = bv; tsum += bv;
    }
    float rinv = 1.0f / tsum;
    for (int k = 0; k < TOPK; ++k) {
      topi[(size_t)t * TOPK + k] = ti[k];
      topv[(size_t)t * TOPK + k] = tv[k] * rinv;
    }
  }
  __syncthreads();
  atomicAdd(&usage[e], p[e]);   // sum over tokens of softmax prob
}

// ---------------------------------------------------------------------------
// Compact (token, weight) lists per expert.
__global__ void moe_scatter(const int* __restrict__ topi,
                            const float* __restrict__ topv,
                            int* __restrict__ cnt,
                            int* __restrict__ tokl, float* __restrict__ wl) {
  int t = blockIdx.x * blockDim.x + threadIdx.x;
  if (t >= T_TOK) return;
#pragma unroll
  for (int k = 0; k < TOPK; ++k) {
    int   e = topi[(size_t)t * TOPK + k];
    float w = topv[(size_t)t * TOPK + k];
    int pos = atomicAdd(&cnt[e], 1);
    if (pos < CAP) {
      tokl[(size_t)e * CAP + pos] = t;
      wl[(size_t)e * CAP + pos]   = w;
    }
  }
}

// ---------------------------------------------------------------------------
// Expert FFN: per (expert, 128-token tile). Fused GEMM1->GELU->GEMM2 with
// the intermediate h slice held in LDS; W1/W2 streamed f32->bf16 in regs.
__global__ void moe_expert_wmma(const unsigned short* __restrict__ xbf,
                                const float* __restrict__ ew1,
                                const float* __restrict__ eb1,
                                const float* __restrict__ ew2,
                                const float* __restrict__ eb2,
                                const int* __restrict__ cnt,
                                const int* __restrict__ tokl,
                                const float* __restrict__ wl,
                                float* __restrict__ out) {
  const int e    = blockIdx.x;
  const int tile = blockIdx.y;
  int nv_total = cnt[e];
  if (nv_total > CAP) nv_total = CAP;
  if (tile * MTILE >= nv_total) return;
  const int n_valid = min(MTILE, nv_total - tile * MTILE);

  extern __shared__ char smem[];
  unsigned short* xs = (unsigned short*)smem;                       // 128x512 bf16
  unsigned short* hs = (unsigned short*)(smem + MTILE * DMODEL * 2);// 128x128 bf16
  int*   stok = (int*)  (smem + MTILE * DMODEL * 2 + MTILE * SLICE * 2);
  float* sw   = (float*)(smem + MTILE * DMODEL * 2 + MTILE * SLICE * 2 + MTILE * 4);

  const float* w1e = ew1 + (size_t)e * DMODEL * DFFN;
  const float* w2e = ew2 + (size_t)e * DFFN * DMODEL;
  const float* b1e = eb1 + (size_t)e * DFFN;
  const float* b2e = eb2 + (size_t)e * DMODEL;

  for (int r = threadIdx.x; r < MTILE; r += blockDim.x) {
    int g = tile * MTILE + r;
    if (g < nv_total) { stok[r] = tokl[(size_t)e * CAP + g]; sw[r] = wl[(size_t)e * CAP + g]; }
    else              { stok[r] = 0;                          sw[r] = 0.0f; }
  }
  __syncthreads();
  // Async-stage gathered token rows (bf16): 64 16-byte chunks per row.
  for (int c = threadIdx.x; c < MTILE * (DMODEL / 8); c += blockDim.x) {
    int r  = c >> 6;              // 64 chunks per 512-col row
    int ko = (c & 63) * 8;        // halfword offset within row
    async_copy_b128(xbf + (size_t)stok[r] * DMODEL + ko,
                    xs + (size_t)r * DMODEL + ko);
  }
  wait_async0();
  __syncthreads();

  const int wave = threadIdx.x >> 5;
  const int lane = threadIdx.x & 31;
  const int half = lane >> 4;
  const int lcol = lane & 15;

  v8f acc2[8][4];   // [M tile][N tile] of the 128x512 output block (per wave: 64 cols)
#pragma unroll
  for (int m = 0; m < 8; ++m)
#pragma unroll
    for (int n = 0; n < 4; ++n) acc2[m][n] = (v8f){0, 0, 0, 0, 0, 0, 0, 0};

  for (int s = 0; s < DFFN / SLICE; ++s) {
    // ---- GEMM1 slice: h[:, s*128 .. +128) ----
    const int ncol = s * SLICE + wave * 16 + lcol;   // global FFN column
    v8f acc1[8];
#pragma unroll
    for (int m = 0; m < 8; ++m) acc1[m] = (v8f){0, 0, 0, 0, 0, 0, 0, 0};

    // warm L2 for the upcoming W2 rows of this slice
    __builtin_prefetch(w2e + (size_t)(s * SLICE) * DMODEL, 0, 0);

    for (int ks = 0; ks < DMODEL; ks += 32) {
      Frag16 b = load_b_f32(w1e, ks, DFFN, ncol, half);
#pragma unroll
      for (int m = 0; m < 8; ++m) {
        Frag16 a = load_a_lds(xs, m * 16 + lcol, DMODEL, ks, half);
        acc1[m] = __builtin_amdgcn_wmma_f32_16x16x32_bf16(
            false, a.v, false, b.v, (short)0, acc1[m], false, false);
      }
    }
    const float bias1 = b1e[ncol];
    const int   lc    = wave * 16 + lcol;            // local column in slice
#pragma unroll
    for (int m = 0; m < 8; ++m) {
#pragma unroll
      for (int r = 0; r < 8; ++r) {
        int row = m * 16 + (half ? r + 8 : r);
        hs[(size_t)row * SLICE + lc] = f2bf(gelu_exact(acc1[m][r] + bias1));
      }
    }
    __syncthreads();

    // ---- GEMM2 partial: out += h_slice @ W2[s*128 .. +128, :] ----
    for (int kk = 0; kk < SLICE; kk += 32) {
      const int kglob = s * SLICE + kk;
      Frag16 b2[4];
#pragma unroll
      for (int nt = 0; nt < 4; ++nt)
        b2[nt] = load_b_f32(w2e, kglob, DMODEL, wave * 64 + nt * 16 + lcol, half);
#pragma unroll
      for (int m = 0; m < 8; ++m) {
        Frag16 a = load_a_lds(hs, m * 16 + lcol, SLICE, kk, half);
#pragma unroll
        for (int nt = 0; nt < 4; ++nt)
          acc2[m][nt] = __builtin_amdgcn_wmma_f32_16x16x32_bf16(
              false, a.v, false, b2[nt].v, (short)0, acc2[m][nt], false, false);
      }
    }
    __syncthreads();   // hs reused next slice
  }

  // ---- epilogue: scale by combine weight, accumulate into out ----
#pragma unroll
  for (int nt = 0; nt < 4; ++nt) {
    const int   n2  = wave * 64 + nt * 16 + lcol;
    const float b2v = b2e[n2];
#pragma unroll
    for (int m = 0; m < 8; ++m) {
#pragma unroll
      for (int r = 0; r < 8; ++r) {
        int row = m * 16 + (half ? r + 8 : r);
        if (row < n_valid) {
          atomicAdd(out + (size_t)stok[row] * DMODEL + n2,
                    sw[row] * (acc2[m][nt][r] + b2v));
        }
      }
    }
  }
}

// ---------------------------------------------------------------------------
__global__ void moe_lb_loss(const float* __restrict__ usage, float* __restrict__ o) {
  __shared__ float red[NEXP];
  int e = threadIdx.x;
  float u = usage[e] * (1.0f / (float)T_TOK);
  float d = u - 1.0f / (float)NEXP;
  red[e] = d * d;
  __syncthreads();
  if (e == 0) {
    float s = 0.0f;
    for (int i = 0; i < NEXP; ++i) s += red[i];
    *o = 0.01f * (s / (float)NEXP);
  }
}

// ---------------------------------------------------------------------------
extern "C" void kernel_launch(void* const* d_in, const int* in_sizes, int n_in,
                              void* d_out, int out_size, void* d_ws, size_t ws_size,
                              hipStream_t stream) {
  (void)in_sizes; (void)n_in; (void)ws_size;
  const float* x   = (const float*)d_in[0];
  const float* gw1 = (const float*)d_in[1];
  const float* gb1 = (const float*)d_in[2];
  const float* gw2 = (const float*)d_in[3];
  const float* gb2 = (const float*)d_in[4];
  const float* ew1 = (const float*)d_in[5];
  const float* eb1 = (const float*)d_in[6];
  const float* ew2 = (const float*)d_in[7];
  const float* eb2 = (const float*)d_in[8];
  float* out = (float*)d_out;

  char* ws = (char*)d_ws;
  unsigned short* h_ws = (unsigned short*)ws;                    // 4 MB bf16 gate hidden
  int*   topi  = (int*)  (ws + 4u * 1024 * 1024);                // 64 KB
  float* topv  = (float*)(ws + 4u * 1024 * 1024 + 64 * 1024);    // 64 KB
  int*   cnt   = (int*)  (ws + 4u * 1024 * 1024 + 128 * 1024);   // 256 B
  float* usage = (float*)(ws + 4u * 1024 * 1024 + 128 * 1024 + 1024); // 256 B
  int*   tokl  = (int*)  (ws + 4u * 1024 * 1024 + 256 * 1024);   // 256 KB
  float* wl    = (float*)(ws + 4u * 1024 * 1024 + 512 * 1024);   // 256 KB
  unsigned short* xbf = (unsigned short*)(ws + 4u * 1024 * 1024 + 768 * 1024); // 2 MB

  // d_out is poisoned; zero it and the small counters.
  moe_zero_f32<<<(out_size + 255) / 256, 256, 0, stream>>>(out, out_size);
  moe_zero_f32<<<2, 256, 0, stream>>>((float*)cnt, 512);  // covers cnt + usage region

  moe_x_to_bf16<<<(T_TOK * DMODEL + 255) / 256, 256, 0, stream>>>(
      x, xbf, T_TOK * DMODEL);

  moe_gate1_wmma<<<dim3(T_TOK / MTILE, HGATE / SLICE), 256,
                   MTILE * DMODEL * 2, stream>>>(xbf, gw1, gb1, h_ws);

  moe_gate2<<<T_TOK, NEXP, 0, stream>>>(h_ws, gw2, gb2, topi, topv, usage);

  moe_scatter<<<(T_TOK + 255) / 256, 256, 0, stream>>>(topi, topv, cnt, tokl, wl);

  const size_t expert_smem = (size_t)MTILE * DMODEL * 2 + (size_t)MTILE * SLICE * 2
                           + (size_t)MTILE * 4 + (size_t)MTILE * 4;
  moe_expert_wmma<<<dim3(NEXP, CAP / MTILE), 256, expert_smem, stream>>>(
      xbf, ew1, eb1, ew2, eb2, cnt, tokl, wl, out);

  moe_lb_loss<<<1, NEXP, 0, stream>>>(usage, out + (out_size - 1));
}
